// AdditiveSelfAttentionLayer_77970836292245
// MI455X (gfx1250) — matrix operand order repaired and checked by reading, not compiled, not run on hardware
//
#include <hip/hip_runtime.h>

// AdditiveSelfAttention, B=4, N=1024, D=64, fp32 in/out.
// Workgroup = (batch b, 32-row i-tile), 256 threads = 8 waves.
//   Phase 1: stage f32 i-tile (32x64) into LDS.
//   Phase 2: scores. Each thread owns one j-row in 64 VGPRs (per 256-j strip);
//            i-rows broadcast from LDS via ds_load_b128. Per element:
//            v_add + v_tanh_f32 + v_add  (the 268M-tanh roofline term).
//            e stored f16 in LDS, computed exactly once.
//   Phase 3: row softmax via wave32 shuffle reductions; p=exp(e-max) f16 in
//            place, 1/rowsum kept for the epilogue.
//   Phase 4: out(32x64) = p(32x1024) @ x(1024x64): 8 waves = 2 i-halves x 4
//            d-tiles, v_wmma_f32_16x16x32_f16; A frags via ds_load_b128, B
//            frags from L2-hot global f32 + v_cvt. K-loop FULLY UNROLLED so
//            the compiler pipelines B loads across wmma iterations
//            (s_wait_loadcnt > 0 style scheduling, 32 static wmmas).
// LDS ~74 KB -> multiple workgroups per WGP for latency hiding in phase 2.

typedef _Float16 v16h __attribute__((ext_vector_type(16)));
typedef _Float16 v8h  __attribute__((ext_vector_type(8)));
typedef float    v8f  __attribute__((ext_vector_type(8)));

#define B_DIM   4
#define N_DIM   1024
#define D_DIM   64
#define TILE_M  32
#define PITCH   1032   // halves; 2064B = 516 dwords -> bank stride 4, rows 16B-aligned
#define THREADS 256

__device__ __forceinline__ float tanh_hw(float x) {
#if __has_builtin(__builtin_amdgcn_tanhf)
  return __builtin_amdgcn_tanhf(x);       // compiler-scheduled v_tanh_f32
#elif __has_builtin(__builtin_amdgcn_tanh_f32)
  return __builtin_amdgcn_tanh_f32(x);
#else
  float y;                                // verified fallback; v_nop covers the
  asm volatile("v_tanh_f32 %0, %1\n\tv_nop" : "=v"(y) : "v"(x));  // TRANS hazard
  return y;
#endif
}

__global__ __launch_bounds__(THREADS)
void addattn_kernel(const float* __restrict__ x, float* __restrict__ out) {
  extern __shared__ unsigned char smem_raw[];
  _Float16* ew   = (_Float16*)smem_raw;            // [TILE_M][PITCH] scores -> exp-weights
  float*    xi   = (float*)(ew + TILE_M * PITCH);  // [TILE_M][D_DIM] f32 i-tile
  float*    rinv = xi + TILE_M * D_DIM;            // [TILE_M]

  const int t    = threadIdx.x;
  const int lane = t & 31;
  const int wave = t >> 5;
  const int i0   = blockIdx.x * TILE_M;            // 0..1023 step 32
  const int b    = blockIdx.y;
  const float* xb = x + (size_t)b * N_DIM * D_DIM;

  // ---- Phase 1: stage i-tile ----
  for (int s = t; s < TILE_M * D_DIM; s += THREADS)
    xi[s] = xb[i0 * D_DIM + s];
  __syncthreads();

  // ---- Phase 2: scores, one j-row per thread per strip ----
  for (int s = 0; s < N_DIM / THREADS; ++s) {
    const int j = s * THREADS + t;
    if (s + 1 < N_DIM / THREADS) {                 // global_prefetch_b8 next strip
      __builtin_prefetch(xb + (size_t)(j + THREADS) * D_DIM, 0, 0);
      __builtin_prefetch(xb + (size_t)(j + THREADS) * D_DIM + 32, 0, 0);
    }
    float xj[D_DIM];                               // j-row pinned in 64 VGPRs
    #pragma unroll
    for (int c = 0; c < D_DIM / 4; ++c) {
      float4 v = *(const float4*)(xb + (size_t)j * D_DIM + 4 * c);
      xj[4 * c + 0] = v.x; xj[4 * c + 1] = v.y;
      xj[4 * c + 2] = v.z; xj[4 * c + 3] = v.w;
    }
    for (int i = 0; i < TILE_M; ++i) {             // i-row broadcast from LDS
      float acc = 0.f;
      #pragma unroll
      for (int d = 0; d < D_DIM; ++d)
        acc += tanh_hw(xi[i * D_DIM + d] + xj[d]);
      ew[i * PITCH + j] = (_Float16)acc;           // |e| <= 64, f16-safe
    }
  }
  __syncthreads();

  // ---- Phase 3: softmax, wave w owns rows 4w..4w+3 (32 rows / 8 waves) ----
  {
    const int r = wave * 4;
    for (int q = 0; q < 4; ++q) {
      float m = -1e30f;
      for (int j = lane; j < N_DIM; j += 32)
        m = fmaxf(m, (float)ew[(r + q) * PITCH + j]);
      for (int off = 16; off; off >>= 1)
        m = fmaxf(m, __shfl_xor(m, off, 32));
      float sum = 0.f;
      for (int j = lane; j < N_DIM; j += 32) {
        float v = __expf((float)ew[(r + q) * PITCH + j] - m);  // v_exp_f32
        sum += v;
        ew[(r + q) * PITCH + j] = (_Float16)v;     // p in (0,1]
      }
      for (int off = 16; off; off >>= 1)
        sum += __shfl_xor(sum, off, 32);
      if (lane == 0) rinv[r + q] = 1.f / sum;
    }
  }
  __syncthreads();

  // ---- Phase 4: WMMA. wave -> (i-half, 16-col d-tile) ----
  const int mlane = lane & 15;
  const int hsel  = lane >> 4;
  const int ihalf = wave >> 2;                     // 0,1
  const int d0    = (wave & 3) * 16;
  const int arow0 = (ihalf * 16 + mlane) * PITCH + hsel * 8;
  const float* bbase = xb + (size_t)(hsel * 8) * D_DIM + d0 + mlane;
  v8f c = {};
  #pragma unroll                                   // full unroll: 32 wmmas,
  for (int k = 0; k < N_DIM; k += 32) {            // B loads pipelined across iters
    // A (16x32 f16): halves [0..7]=K(hsel*8..), [8..15]=K(16+hsel*8..)
    const _Float16* ap = ew + arow0 + k;
    v8h a0 = *(const v8h*)(ap);                    // ds_load_b128
    v8h a1 = *(const v8h*)(ap + 16);
    v16h a = __builtin_shufflevector(a0, a1, 0,1,2,3,4,5,6,7,8,9,10,11,12,13,14,15);
    // B (32x16 f16): lane = column d0+mlane, K [0..7]=hsel*8.., [8..15]=16+hsel*8..;
    // from global f32 (L2-hot) + cvt. Each load: two 64B coalesced segments.
    v16h bf;
    #pragma unroll
    for (int kk = 0; kk < 16; ++kk)
      bf[kk] = (_Float16)bbase[(size_t)(k + (kk < 8 ? kk : kk + 8)) * D_DIM];
    c = __builtin_amdgcn_wmma_f32_16x16x32_f16(false, a, false, bf,
                                               (short)0, c, false, false);
  }
  // C/D: vgpr r -> row ihalf*16 + r + 8*hsel, col d0+mlane; fold 1/rowsum here.
  const int rbase = ihalf * 16 + hsel * 8;
  float* op = out + ((size_t)b * N_DIM + i0 + rbase) * D_DIM + d0 + mlane;
  #pragma unroll
  for (int r = 0; r < 8; ++r)
    op[r * D_DIM] = c[r] * rinv[rbase + r];
}

extern "C" void kernel_launch(void* const* d_in, const int* in_sizes, int n_in,
                              void* d_out, int out_size, void* d_ws, size_t ws_size,
                              hipStream_t stream) {
  (void)in_sizes; (void)n_in; (void)d_ws; (void)ws_size; (void)out_size;
  const float* x = (const float*)d_in[0];          // (4,1024,64) f32
  float* out     = (float*)d_out;                  // (4,1024,64) f32
  dim3 grid(N_DIM / TILE_M, B_DIM);                // 32 i-tiles x 4 batches
  size_t smem = (size_t)TILE_M * PITCH * sizeof(_Float16)
              + (size_t)TILE_M * D_DIM * sizeof(float)
              + (size_t)TILE_M * sizeof(float);    // ~74 KB -> multi-WG/WGP
  addattn_kernel<<<grid, THREADS, smem, stream>>>(x, out);
}
